// ContextAttention_21457656611319
// MI455X (gfx1250) — compile-verified, matrix operand order checked by reading, not run on hardware
//
#include <hip/hip_runtime.h>

typedef _Float16 h8  __attribute__((ext_vector_type(8)));
typedef _Float16 h16 __attribute__((ext_vector_type(16)));
typedef float    f8  __attribute__((ext_vector_type(8)));
typedef float    f4  __attribute__((ext_vector_type(4)));
typedef unsigned u32x4 __attribute__((ext_vector_type(4)));
typedef int      i32x4 __attribute__((ext_vector_type(4)));
typedef int      i32x8 __attribute__((ext_vector_type(8)));

constexpr int BN = 4;      // batch
constexpr int CC = 128;    // channels
constexpr int LL = 4096;   // H*W
constexpr int CP = CC + 8; // padded LDS row (halves) -> avoids 64-bank conflicts

static __device__ __forceinline__ f8 fzero() {
  f8 z;
#pragma unroll
  for (int j = 0; j < 8; ++j) z[j] = 0.0f;
  return z;
}

// Load one 16-bit WMMA fragment for this lane: 8 halves at p, 8 at p+16.
static __device__ __forceinline__ h16 loadfrag(const _Float16* p) {
  h16 r;
  *(h8*)&r       = *(const h8*)p;
  *((h8*)&r + 1) = *(const h8*)(p + 16);
  return r;
}

static __device__ __forceinline__ f8 wmma_f16(h16 a, h16 b, f8 c) {
  return __builtin_amdgcn_wmma_f32_16x16x32_f16(false, a, false, b,
                                                (short)0, c, false, false);
}

// ---------------------------------------------------------------------------
// Kernel 1: q,k,v = W @ x + b  (1x1 conv). One block = one batch n and a
// 64-pixel tile. Outputs f16:
//   QT[n][m][c]  (scaled by 1/sqrt(C)),  KT[n][l][c],  V[n][c][l]
// ---------------------------------------------------------------------------
__global__ __launch_bounds__(256) void qkv_kernel(
    const float* __restrict__ x,
    const float* __restrict__ Wq, const float* __restrict__ bq,
    const float* __restrict__ Wk, const float* __restrict__ bk,
    const float* __restrict__ Wv, const float* __restrict__ bv,
    _Float16* __restrict__ QT, _Float16* __restrict__ KT,
    _Float16* __restrict__ V)
{
  // [0, 34816): Wl [128][CP] f16 ; [34816, ...): xT [64][CP] f16, later reused
  // as vstage [128][72] f16 (overlap is barrier-protected).
  __shared__ char smem[53248];
  _Float16* Wl  = (_Float16*)smem;
  _Float16* xT  = (_Float16*)(smem + 34816);
  _Float16* vst = (_Float16*)(smem + 34816);

  const int tid  = threadIdx.x;
  const int lane = tid & 31;
  const int wave = tid >> 5;
  const int n    = blockIdx.x >> 6;        // LL/64 = 64 tiles per batch
  const int mb   = (blockIdx.x & 63) << 6;

  const int row  = lane & 15;              // fragment row/col within tile
  const int koff = (lane >> 4) << 3;       // per-lane K offset {0,8}

  // stage xT[m][c] (f32 -> f16), coalesced reads over m
  for (int idx = tid; idx < CC * 64; idx += 256) {
    int c = idx >> 6, m = idx & 63;
    xT[m * CP + c] = (_Float16)x[(n * CC + c) * LL + mb + m];
  }

  const float* Ws[3] = {Wq, Wk, Wv};
  const float* bs[3] = {bq, bk, bv};
  const float  scale = 0.08838834764831845f;  // 1/sqrt(128)

  for (int wmat = 0; wmat < 3; ++wmat) {
    __syncthreads();  // xT staged (iter 0) / Wl no longer read (iters 1,2)
    const float* Wsrc = Ws[wmat];
    for (int idx = tid; idx < CC * CC; idx += 256) {
      int o = idx >> 7, c = idx & 127;
      Wl[o * CP + c] = (_Float16)Wsrc[idx];
    }
    __syncthreads();

    const int obase = wave << 4;  // this wave's 16-row output channel block
    f8 acc[4];
#pragma unroll
    for (int mt = 0; mt < 4; ++mt) acc[mt] = fzero();

#pragma unroll
    for (int cb = 0; cb < CC; cb += 32) {
      h16 a = loadfrag(&Wl[(obase + row) * CP + cb + koff]);
#pragma unroll
      for (int mt = 0; mt < 4; ++mt) {
        h16 b = loadfrag(&xT[(mt * 16 + row) * CP + cb + koff]);
        acc[mt] = wmma_f16(a, b, acc[mt]);
      }
    }

    // bias for this lane's 8 rows (contiguous channels obase+koff..+7)
    const float* bias = bs[wmat];
    float bv8[8];
    {
      f4 t0 = *(const f4*)(bias + obase + koff);
      f4 t1 = *(const f4*)(bias + obase + koff + 4);
#pragma unroll
      for (int j = 0; j < 4; ++j) { bv8[j] = t0[j]; bv8[4 + j] = t1[j]; }
    }

    if (wmat < 2) {
      _Float16* dst = (wmat == 0) ? QT : KT;
      float scl = (wmat == 0) ? scale : 1.0f;
#pragma unroll
      for (int mt = 0; mt < 4; ++mt) {
        int m = mb + mt * 16 + row;
        h8 o8;
#pragma unroll
        for (int j = 0; j < 8; ++j)
          o8[j] = (_Float16)((acc[mt][j] + bv8[j]) * scl);
        *(h8*)&dst[(n * LL + m) * CC + obase + koff] = o8;  // 16B store
      }
    } else {
      __syncthreads();  // xT dead now; vstage overlaps it
#pragma unroll
      for (int mt = 0; mt < 4; ++mt) {
        int m = mt * 16 + row;
#pragma unroll
        for (int j = 0; j < 8; ++j)
          vst[(obase + koff + j) * 72 + m] = (_Float16)(acc[mt][j] + bv8[j]);
      }
      __syncthreads();
      // coalesced copy-out: V[n][c][mb..mb+64)
      for (int idx = tid; idx < (CC * 64) / 16; idx += 256) {
        int c = idx >> 2, ch = (idx & 3) << 4;
        h8 a0 = *(const h8*)&vst[c * 72 + ch];
        h8 a1 = *(const h8*)&vst[c * 72 + ch + 8];
        *(h8*)&V[(n * CC + c) * LL + mb + ch]     = a0;
        *(h8*)&V[(n * CC + c) * LL + mb + ch + 8] = a1;
      }
    }
  }
}

// ---------------------------------------------------------------------------
// Kernel 2: per-row softmax statistics. Row l: max_m s[l,m] and
// 1/sum_m exp(s-max). Wave owns a 16-row l-strip; K^T fragments stay
// register-resident across the whole m loop (KT read once per wave).
// ---------------------------------------------------------------------------
__global__ __launch_bounds__(256) void stats_kernel(
    const _Float16* __restrict__ QT, const _Float16* __restrict__ KT,
    float* __restrict__ rowmax, float* __restrict__ rowinv)
{
  const int tid  = threadIdx.x;
  const int lane = tid & 31;
  const int wave = tid >> 5;
  const int n    = blockIdx.x >> 5;                       // LL/128 = 32 tiles
  const int lb   = ((blockIdx.x & 31) << 7) + (wave << 4);
  const int row  = lane & 15;
  const int koff = (lane >> 4) << 3;

  h16 a[4];
#pragma unroll
  for (int cb = 0; cb < 4; ++cb)
    a[cb] = loadfrag(&KT[(n * LL + lb + row) * CC + cb * 32 + koff]);

  // pass 1: row max
  f8 rmax;
#pragma unroll
  for (int j = 0; j < 8; ++j) rmax[j] = -3.0e38f;
  for (int m = 0; m < LL; m += 16) {
    const _Float16* qcol = &QT[(n * LL + m + row) * CC + koff];
    f8 s = fzero();
#pragma unroll
    for (int cb = 0; cb < 4; ++cb)
      s = wmma_f16(a[cb], loadfrag(qcol + cb * 32), s);
#pragma unroll
    for (int j = 0; j < 8; ++j) rmax[j] = fmaxf(rmax[j], s[j]);
  }
#pragma unroll
  for (int mask = 1; mask < 16; mask <<= 1)
#pragma unroll
    for (int j = 0; j < 8; ++j)
      rmax[j] = fmaxf(rmax[j], __shfl_xor(rmax[j], mask, 32));

  // pass 2: sum of exp(s - max)
  f8 rsum = fzero();
  for (int m = 0; m < LL; m += 16) {
    const _Float16* qcol = &QT[(n * LL + m + row) * CC + koff];
    f8 s = fzero();
#pragma unroll
    for (int cb = 0; cb < 4; ++cb)
      s = wmma_f16(a[cb], loadfrag(qcol + cb * 32), s);
#pragma unroll
    for (int j = 0; j < 8; ++j) rsum[j] += __expf(s[j] - rmax[j]);
  }
#pragma unroll
  for (int mask = 1; mask < 16; mask <<= 1)
#pragma unroll
    for (int j = 0; j < 8; ++j) rsum[j] += __shfl_xor(rsum[j], mask, 32);

  if ((lane & 15) == 0) {
    int base = n * LL + lb + koff;
#pragma unroll
    for (int j = 0; j < 8; ++j) {
      rowmax[base + j] = rmax[j];
      rowinv[base + j] = 1.0f / rsum[j];
    }
  }
}

// ---------------------------------------------------------------------------
// Kernel 3: attn[c,m] = sum_l V[c,l] * exp(s[l,m]-max[l]) * inv[l]; out = x+attn.
// Q tile staged by the Tensor Data Mover (D# with HW row padding), P tile
// double-buffered in LDS (one barrier per 32-row l step). 8 WMMAs/wave/step.
// ---------------------------------------------------------------------------
__global__ __launch_bounds__(256) void attn_kernel(
    const float* __restrict__ x,
    const _Float16* __restrict__ QT, const _Float16* __restrict__ KT,
    const _Float16* __restrict__ V,
    const float* __restrict__ rowmax, const float* __restrict__ rowinv,
    float* __restrict__ out)
{
  __shared__ _Float16 Qs[64 * CP];      // Q tile [m][c], CP-padded rows
  __shared__ _Float16 Ps[2][64 * 40];   // P tile [m][l(32)+pad], double-buffered

  const int tid  = threadIdx.x;
  const int lane = tid & 31;
  const int wave = tid >> 5;
  const int n    = blockIdx.x >> 6;
  const int mb   = (blockIdx.x & 63) << 6;
  const int row  = lane & 15;
  const int koff = (lane >> 4) << 3;

  // ---- stage Q tile: TDM 2D tile load with hardware LDS row padding ----
#if defined(__AMDGCN__) && __has_builtin(__builtin_amdgcn_tensor_load_to_lds)
  if (wave == 0) {
    unsigned long long ga =
        (unsigned long long)(const void*)&QT[(size_t)(n * LL + mb) * CC];
    u32x4 g0;
    g0[0] = 1u;                                   // count=1 (valid descriptor)
    g0[1] = (unsigned)(unsigned long long)(void*)Qs;  // lds_addr (bytes)
    g0[2] = (unsigned)ga;                         // global_addr[31:0]
    g0[3] = (unsigned)(ga >> 32) | (2u << 30);    // addr[56:32] | type=2
    i32x8 g1;
    // data_size=1 (2B) | pad_enable | pad_interval=5 (64 DW = 128 halves)
    // | pad_amount=3 (4 DW = 8 halves)
    g1[0] = (1 << 16) | (1 << 20) | (5 << 22) | (3 << 25);
    g1[1] = (int)(128u << 16);        // tensor_dim0[15:0] = 128 (c)
    g1[2] = (int)(0x1000u << 16);     // tensor_dim1[15:0] = 4096 (rows)
    g1[3] = (int)(128u << 16);        // tile_dim0 = 128
    g1[4] = 64;                       // tile_dim1 = 64, tile_dim2 = 0
    g1[5] = 128;                      // tensor_dim0_stride = 128
    g1[6] = 0;
    g1[7] = 0;
    i32x4 gz;
#pragma unroll
    for (int j = 0; j < 4; ++j) gz[j] = 0;
#if __clang_major__ >= 23
    i32x8 g4z;
#pragma unroll
    for (int j = 0; j < 8; ++j) g4z[j] = 0;
    __builtin_amdgcn_tensor_load_to_lds(g0, g1, gz, gz, g4z, 0);
#else
    __builtin_amdgcn_tensor_load_to_lds(g0, g1, gz, gz, 0);
#endif
    __builtin_amdgcn_s_wait_tensorcnt(0);
  }
#else
  for (int idx = tid; idx < (64 * CC) / 8; idx += 256) {
    int m = idx >> 4, c = (idx & 15) << 3;
    *(h8*)&Qs[m * CP + c] = *(const h8*)&QT[(n * LL + mb + m) * CC + c];
  }
#endif
  __syncthreads();

  f8 accO[4];
#pragma unroll
  for (int mt = 0; mt < 4; ++mt) accO[mt] = fzero();

  const int lt  = (wave & 1) << 4;   // S-tile l offset within 32-row step
  const int mts = (wave >> 1) << 4;  // S-tile m offset within 64-col tile
  const int cw  = wave << 4;         // PV channel strip

  int buf = 0;
  for (int ls = 0; ls < LL; ls += 32) {
    // ---- S tile (16l x 16m), K-loop over c ----
    f8 s = fzero();
    const _Float16* arow = &KT[(n * LL + ls + lt + row) * CC + koff];
    const _Float16* brow = &Qs[(mts + row) * CP + koff];
    __builtin_prefetch(arow + 32 * CC, 0, 0);  // next l-step K rows -> near
#pragma unroll
    for (int cb = 0; cb < 4; ++cb)
      s = wmma_f16(loadfrag(arow + cb * 32), loadfrag(brow + cb * 32), s);

    // ---- p = exp(s - rowmax) * rowinv, packed write into Ps[buf][m][l] ----
    const int lrow = n * LL + ls + lt + koff;  // 8 consecutive rows per lane
    f4 rm0 = *(const f4*)&rowmax[lrow];
    f4 rm1 = *(const f4*)&rowmax[lrow + 4];
    f4 ri0 = *(const f4*)&rowinv[lrow];
    f4 ri1 = *(const f4*)&rowinv[lrow + 4];
    h8 p;
#pragma unroll
    for (int j = 0; j < 4; ++j)
      p[j] = (_Float16)(__expf(s[j] - rm0[j]) * ri0[j]);
#pragma unroll
    for (int j = 0; j < 4; ++j)
      p[4 + j] = (_Float16)(__expf(s[4 + j] - rm1[j]) * ri1[j]);
    *(h8*)&Ps[buf][(mts + row) * 40 + lt + koff] = p;  // one b128 LDS store
    __syncthreads();  // single barrier: double buffer protects reuse

    // ---- accO += V(16c x 32l) * P(32l x 16m) ----
    const _Float16* vrow = &V[(n * CC + cw + row) * LL + ls + koff];
    __builtin_prefetch(vrow + 32, 0, 0);       // next l-step V fragment
    h16 av = loadfrag(vrow);
#pragma unroll
    for (int mt = 0; mt < 4; ++mt) {
      h16 bp = loadfrag(&Ps[buf][(mt * 16 + row) * 40 + koff]);
      accO[mt] = wmma_f16(av, bp, accO[mt]);
    }
    buf ^= 1;
  }

  // epilogue: out = x + attn
#pragma unroll
  for (int mt = 0; mt < 4; ++mt) {
    int m = mb + mt * 16 + row;
#pragma unroll
    for (int j = 0; j < 8; ++j) {
      int c  = cw + koff + j;
      int gi = (n * CC + c) * LL + m;
      out[gi] = x[gi] + accO[mt][j];
    }
  }
}

// ---------------------------------------------------------------------------
extern "C" void kernel_launch(void* const* d_in, const int* in_sizes, int n_in,
                              void* d_out, int out_size, void* d_ws,
                              size_t ws_size, hipStream_t stream) {
  const float* x  = (const float*)d_in[0];
  const float* Wq = (const float*)d_in[1];
  const float* bq = (const float*)d_in[2];
  const float* Wk = (const float*)d_in[3];
  const float* bk = (const float*)d_in[4];
  const float* Wv = (const float*)d_in[5];
  const float* bv = (const float*)d_in[6];
  float* out = (float*)d_out;

  char* ws = (char*)d_ws;
  const size_t tensz = (size_t)BN * LL * CC * sizeof(_Float16);  // 4 MB each
  _Float16* QT = (_Float16*)ws;
  _Float16* KT = (_Float16*)(ws + tensz);
  _Float16* V  = (_Float16*)(ws + 2 * tensz);
  float* rmax  = (float*)(ws + 3 * tensz);
  float* rinv  = rmax + BN * LL;

  qkv_kernel<<<BN * (LL / 64), 256, 0, stream>>>(x, Wq, bq, Wk, bk, Wv, bv,
                                                 QT, KT, V);
  stats_kernel<<<BN * (LL / 128), 256, 0, stream>>>(QT, KT, rmax, rinv);
  attn_kernel<<<BN * (LL / 64), 256, 0, stream>>>(x, QT, KT, V, rmax, rinv,
                                                  out);
}
